// NemotronHMamba2MixerSimple_83416854823038
// MI455X (gfx1250) — matrix operand order repaired and checked by reading, not compiled, not run on hardware
//
#include <hip/hip_runtime.h>
#include <stdint.h>

// ---------------------------------------------------------------------------
// Types
// ---------------------------------------------------------------------------
typedef __bf16 bf16_t;
typedef __attribute__((ext_vector_type(8)))  __bf16   v8bf;
typedef __attribute__((ext_vector_type(16))) __bf16   v16bf;
typedef __attribute__((ext_vector_type(8)))  float    v8f;
typedef __attribute__((ext_vector_type(4)))  uint32_t v4u;
typedef __attribute__((ext_vector_type(8)))  int      v8i;
typedef __attribute__((ext_vector_type(4)))  int      v4i;

__device__ __forceinline__ bf16_t f2bf(float x) { return (bf16_t)x; }

// Tensor Data Mover available? (gfx1250 device pass only)
#if defined(__gfx1250__) && __has_builtin(__builtin_amdgcn_tensor_load_to_lds)
#define USE_TDM 1
#endif

// ---------------------------------------------------------------------------
// Problem constants (from the reference)
// ---------------------------------------------------------------------------
#define HIDDEN_  2048
#define INTER_   4096
#define GN_      1024
#define NHEADS_  64
#define HEADDIM_ 64
#define NGROUPS_ 8
#define NSTATE_  128
#define CONVD_   6144                     // INTER + 2*GN
#define SEQ_     2048
#define BATCH_   2
#define MROWS_   (BATCH_ * SEQ_)          // 4096
#define NTOT_    (INTER_ + CONVD_ + NHEADS_)  // 10304: [gate|x|B|C|dt]
#define XBC_OFF_ INTER_                   // col 4096 in Y_all
#define DT_OFF_  (INTER_ + CONVD_)        // col 10240 in Y_all

// ---------------------------------------------------------------------------
// f32 -> bf16 elementwise cast
// ---------------------------------------------------------------------------
__global__ __launch_bounds__(256)
void cast_bf16_kernel(const float* __restrict__ s, bf16_t* __restrict__ d, int n) {
  int i = blockIdx.x * 256 + threadIdx.x;
  if (i < n) d[i] = f2bf(s[i]);
}

// ---------------------------------------------------------------------------
// Tiled transpose + bf16 pack: dst[(ndstoff+n)*Kdst + k] = bf16(src[k*Nsrc + n])
// ---------------------------------------------------------------------------
__global__ __launch_bounds__(256)
void transpose_pack_kernel(const float* __restrict__ src, bf16_t* __restrict__ dst,
                           int K, int Nsrc, int ndstoff, int Kdst) {
  __shared__ float tile[32][33];
  const int nb = blockIdx.x * 32;
  const int kb = blockIdx.y * 32;
  const int tx = threadIdx.x;  // 0..31
  const int ty = threadIdx.y;  // 0..7
#pragma unroll
  for (int i = ty; i < 32; i += 8) {
    int k = kb + i, n = nb + tx;
    tile[i][tx] = (k < K && n < Nsrc) ? src[(size_t)k * Nsrc + n] : 0.f;
  }
  __syncthreads();
#pragma unroll
  for (int i = ty; i < 32; i += 8) {
    int n = nb + i, k = kb + tx;
    if (n < Nsrc && k < K)
      dst[(size_t)(ndstoff + n) * Kdst + k] = f2bf(tile[tx][i]);
  }
}

// ---------------------------------------------------------------------------
// TDM tile DMA: global [tile_rows x 64B] (row stride = stride_bytes) -> LDS,
// with HW padding 16B after every 64B => LDS row stride 80B (= LSTR_ bf16).
// Rows beyond rows_rem read as zero (ragged N tiles).
// ---------------------------------------------------------------------------
#ifdef USE_TDM
__device__ __forceinline__ void tdm_load_tile(uint32_t lds_off, uint64_t gaddr,
                                              uint32_t rows_rem, uint32_t stride_bytes,
                                              uint32_t tile_rows) {
  v4u g0;
  g0[0] = 1u;                                      // count=1, user descriptor
  g0[1] = lds_off;                                 // D#.lds_addr (bytes)
  g0[2] = (uint32_t)gaddr;                         // global_addr[31:0]
  g0[3] = ((uint32_t)(gaddr >> 32) & 0x01ffffffu)  // global_addr[56:32]
          | (2u << 30);                            // type = 2 ("image")
  v8i g1;
  g1[0] = (int)((0u << 16)       /* data_size = 1 byte            */
              | (1u << 20)       /* pad_enable                    */
              | (3u << 22)       /* pad_interval: 16 DW = 64 B    */
              | (3u << 25));     /* pad_amount:    4 DW = 16 B    */
  g1[1] = (int)((64u & 0xffffu) << 16);                            // tensor_dim0 lo
  g1[2] = (int)((rows_rem & 0xffffu) << 16);                       // td0 hi=0 | tensor_dim1 lo
  g1[3] = (int)((rows_rem >> 16) | (64u << 16));                   // td1 hi | tile_dim0 = 64B
  g1[4] = (int)(tile_rows & 0xffffu);                              // tile_dim1 | tile_dim2=0
  g1[5] = (int)stride_bytes;                                       // tensor_dim0_stride lo
  g1[6] = 0;                                                       // stride0 hi | stride1 lo
  g1[7] = 0;                                                       // stride1 hi
  v4i z4 = {0, 0, 0, 0};
#if __clang_major__ >= 23
  v8i z8 = {0, 0, 0, 0, 0, 0, 0, 0};
  __builtin_amdgcn_tensor_load_to_lds(g0, g1, z4, z4, z8, 0);
#else
  __builtin_amdgcn_tensor_load_to_lds(g0, g1, z4, z4, 0);
#endif
}
#endif

// ---------------------------------------------------------------------------
// WMMA bf16 GEMM: C[M,N] = A[M,K] * Bt[N,K]^T   (A row-major, B pre-transposed)
// 256 threads = 8 waves; block tile 128x128, BK=32; wave tile 32x64 (8 wmma acc)
// Staging: TDM tensor_load_to_lds (wave0 -> A tile, wave1 -> B tile) when
// available, else vector loads + ds_store_b128.
// ---------------------------------------------------------------------------
#define BM_ 128
#define BN_ 128
#define BK_ 32
#define LSTR_ 40   // bf16 elements per LDS row (80B, 16B aligned, conflict-free)

__global__ __launch_bounds__(256, 2)
void wmma_gemm_bf16_kernel(const bf16_t* __restrict__ A, const bf16_t* __restrict__ Bt,
                           float* __restrict__ C, int M, int N, int K) {
  __shared__ __align__(16) bf16_t sA[2][BM_ * LSTR_];
  __shared__ __align__(16) bf16_t sB[2][BN_ * LSTR_];

  const int tid  = threadIdx.x;
  const int bm   = blockIdx.y * BM_;
  const int bn   = blockIdx.x * BN_;
  const int wave = tid >> 5;
  const int lane = tid & 31;
  const int half = lane >> 4;   // 0/1: which 16-lane half
  const int l16  = lane & 15;
  const int wm   = (wave >> 1) * 32;  // wave M origin within block tile
  const int wn   = (wave & 1) * 64;   // wave N origin within block tile

  v8f acc[2][4];
#pragma unroll
  for (int i = 0; i < 2; ++i)
#pragma unroll
    for (int j = 0; j < 4; ++j)
#pragma unroll
      for (int e = 0; e < 8; ++e) acc[i][j][e] = 0.f;

  const int nk = K / BK_;

#ifdef USE_TDM
  // --- TDM staging: wave0 DMAs the A tile, wave1 the B tile. TENSORcnt is
  // per-wave, so the pre-barrier wait is unconditional (no-op elsewhere). ---
  const uint32_t ldsA0   = (uint32_t)(uintptr_t)&sA[0][0];
  const uint32_t ldsB0   = (uint32_t)(uintptr_t)&sB[0][0];
  const uint64_t gA0     = (uint64_t)(uintptr_t)A  + (uint64_t)bm * (uint32_t)K * 2u;
  const uint64_t gB0     = (uint64_t)(uintptr_t)Bt + (uint64_t)bn * (uint32_t)K * 2u;
  const uint32_t remA    = (uint32_t)(M - bm);
  const uint32_t remB    = (uint32_t)(N - bn);
  const uint32_t strideK = (uint32_t)K * 2u;

#define GEMM_STAGE(P, KK)                                                        \
  {                                                                              \
    if (wave == 0)                                                               \
      tdm_load_tile(ldsA0 + (uint32_t)(P) * (uint32_t)(BM_ * LSTR_ * 2),         \
                    gA0 + (uint32_t)(KK) * 2u, remA, strideK, BM_);              \
    else if (wave == 1)                                                          \
      tdm_load_tile(ldsB0 + (uint32_t)(P) * (uint32_t)(BN_ * LSTR_ * 2),         \
                    gB0 + (uint32_t)(KK) * 2u, remB, strideK, BN_);              \
  }
#define GEMM_WAITSTAGE() __builtin_amdgcn_s_wait_tensorcnt(0);

#else
  // --- fallback staging: 256 threads, 32B each per tile, via ds_store_b128 ---
  const int  srow   = tid >> 1;          // staging: row 0..127
  const int  sseg   = (tid & 1) * 16;    // staging: first/second 16 bf16 of BK
  const int  brow   = bn + srow;
  const bool bvalid = brow < N;

#define GEMM_STAGE(P, KK)                                                        \
  {                                                                              \
    size_t ka = (size_t)(bm + srow) * K + (size_t)(KK) + sseg;                   \
    v4u a0 = *(const v4u*)(A + ka);                                              \
    v4u a1 = *(const v4u*)(A + ka + 8);                                          \
    *(v4u*)&sA[P][srow * LSTR_ + sseg]     = a0;                                 \
    *(v4u*)&sA[P][srow * LSTR_ + sseg + 8] = a1;                                 \
    v4u b0, b1;                                                                  \
    if (bvalid) {                                                                \
      size_t kb = (size_t)brow * K + (size_t)(KK) + sseg;                        \
      b0 = *(const v4u*)(Bt + kb);                                               \
      b1 = *(const v4u*)(Bt + kb + 8);                                           \
      __builtin_prefetch(Bt + kb + 2 * BK_, 0, 1);                               \
    } else {                                                                     \
      for (int e = 0; e < 4; ++e) { b0[e] = 0u; b1[e] = 0u; }                    \
    }                                                                            \
    *(v4u*)&sB[P][srow * LSTR_ + sseg]     = b0;                                 \
    *(v4u*)&sB[P][srow * LSTR_ + sseg + 8] = b1;                                 \
    __builtin_prefetch(A + ka + 2 * BK_, 0, 1);                                  \
  }
#define GEMM_WAITSTAGE()
#endif

  // Fragment loads per ISA 16-bit WMMA layouts + 8 WMMAs on buffer P.
#define GEMM_COMPUTE(P)                                                          \
  {                                                                              \
    const bf16_t* pa = &sA[P][0];                                                \
    const bf16_t* pb = &sB[P][0];                                                \
    v16bf afr[2], bfr[4];                                                        \
    _Pragma("unroll")                                                            \
    for (int i = 0; i < 2; ++i) {                                                \
      const bf16_t* p = pa + (wm + i * 16 + l16) * LSTR_ + half * 8;             \
      v8bf lo = *(const v8bf*)p;                                                 \
      v8bf hi = *(const v8bf*)(p + 16);                                          \
      afr[i] = __builtin_shufflevector(lo, hi, 0, 1, 2, 3, 4, 5, 6, 7,           \
                                       8, 9, 10, 11, 12, 13, 14, 15);            \
    }                                                                            \
    _Pragma("unroll")                                                            \
    for (int j = 0; j < 4; ++j) {                                                \
      const bf16_t* p = pb + (wn + j * 16 + l16) * LSTR_ + half * 8;             \
      v8bf lo = *(const v8bf*)p;                                                 \
      v8bf hi = *(const v8bf*)(p + 16);                                          \
      bfr[j] = __builtin_shufflevector(lo, hi, 0, 1, 2, 3, 4, 5, 6, 7,           \
                                       8, 9, 10, 11, 12, 13, 14, 15);            \
    }                                                                            \
    _Pragma("unroll")                                                            \
    for (int i = 0; i < 2; ++i)                                                  \
      _Pragma("unroll")                                                          \
      for (int j = 0; j < 4; ++j)                                                \
        acc[i][j] = __builtin_amdgcn_wmma_f32_16x16x32_bf16(                     \
            false, afr[i], false, bfr[j], (short)0, acc[i][j], false, false);    \
  }

  GEMM_STAGE(0, 0);
  GEMM_WAITSTAGE();
  __syncthreads();
  int buf = 0;
  for (int kt = 0; kt < nk - 1; ++kt) {        // steady state: branch-free body
    GEMM_STAGE(buf ^ 1, (kt + 1) * BK_);
    GEMM_COMPUTE(buf);
    GEMM_WAITSTAGE();
    __syncthreads();
    buf ^= 1;
  }
  GEMM_COMPUTE(buf);                           // peeled last K-step

  // C/D layout: lane -> (M = e + 8*half, N = l16) within each 16x16 tile
#pragma unroll
  for (int i = 0; i < 2; ++i)
#pragma unroll
    for (int j = 0; j < 4; ++j) {
      int col = bn + wn + j * 16 + l16;
      if (col < N) {
        int row0 = bm + wm + i * 16 + half * 8;
#pragma unroll
        for (int e = 0; e < 8; ++e)
          C[(size_t)(row0 + e) * N + col] = acc[i][j][e];
      }
    }
#undef GEMM_STAGE
#undef GEMM_WAITSTAGE
#undef GEMM_COMPUTE
}

// ---------------------------------------------------------------------------
// Causal depthwise conv1d (K=4) + SiLU over the xBC slice of Y_all
// ---------------------------------------------------------------------------
__global__ __launch_bounds__(256)
void conv_silu_kernel(const float* __restrict__ yall, const float* __restrict__ cw,
                      const float* __restrict__ cb, float* __restrict__ xbc) {
  int idx = blockIdx.x * 256 + threadIdx.x;
  if (idx >= MROWS_ * CONVD_) return;
  int r = idx / CONVD_;
  int c = idx - r * CONVD_;
  int s = r & (SEQ_ - 1);   // SEQ_ power of two
  float acc = cb[c];
  const float w0 = cw[c * 4 + 0], w1 = cw[c * 4 + 1];
  const float w2 = cw[c * 4 + 2], w3 = cw[c * 4 + 3];
  if (s >= 3) acc = fmaf(yall[(size_t)(r - 3) * NTOT_ + XBC_OFF_ + c], w0, acc);
  if (s >= 2) acc = fmaf(yall[(size_t)(r - 2) * NTOT_ + XBC_OFF_ + c], w1, acc);
  if (s >= 1) acc = fmaf(yall[(size_t)(r - 1) * NTOT_ + XBC_OFF_ + c], w2, acc);
  acc = fmaf(yall[(size_t)r * NTOT_ + XBC_OFF_ + c], w3, acc);
  xbc[(size_t)r * CONVD_ + c] = acc / (1.f + __expf(-acc));  // silu
}

// ---------------------------------------------------------------------------
// Selective SSM scan: one block per (b,h); state [64x128] in registers
// (32 floats/thread x 256 threads); 4 lanes share one d, shfl-reduce over n
// ---------------------------------------------------------------------------
__global__ __launch_bounds__(256)
void ssm_scan_kernel(const float* __restrict__ xbc, const float* __restrict__ yall,
                     const float* __restrict__ dt_bias, const float* __restrict__ A_log,
                     const float* __restrict__ Dvec, float* __restrict__ out) {
  const int blk = blockIdx.x;
  const int b = blk >> 6;       // / NHEADS_
  const int h = blk & 63;
  const int g = h >> 3;         // head group (hpg = 8)
  const int t = threadIdx.x;
  const int d = t >> 2;         // 0..63
  const int np = t & 3;         // n-partition: 32 states each
  const float Ah    = -__expf(A_log[h]);
  const float dbias = dt_bias[h];
  const float Dh    = Dvec[h];

  __shared__ float sx[2][HEADDIM_];
  __shared__ float sBm[2][NSTATE_];
  __shared__ float sCm[2][NSTATE_];

  float st[32];
#pragma unroll
  for (int j = 0; j < 32; ++j) st[j] = 0.f;

  const size_t rowbase = (size_t)b * SEQ_;

#define SCAN_STAGE(P, S2)                                                   \
  {                                                                         \
    const float* xr = xbc + (rowbase + (S2)) * CONVD_;                      \
    for (int ii = t; ii < 320; ii += 256) {                                 \
      if (ii < 64)        sx[P][ii]       = xr[h * 64 + ii];                \
      else if (ii < 192)  sBm[P][ii - 64] = xr[INTER_ + g * 128 + ii - 64]; \
      else                sCm[P][ii - 192] = xr[INTER_ + GN_ + g * 128 + ii - 192]; \
    }                                                                       \
  }

  SCAN_STAGE(0, 0);
  __syncthreads();
  int p = 0;
  for (int s = 0; s < SEQ_; ++s) {
    if (s + 1 < SEQ_) SCAN_STAGE(p ^ 1, s + 1);

    float dtraw = yall[(rowbase + s) * NTOT_ + DT_OFF_ + h] + dbias;
    float dt = (dtraw > 20.f) ? dtraw : log1pf(__expf(dtraw));  // softplus
    float dA = __expf(dt * Ah);
    float xd = sx[p][d];
    float dtx = dt * xd;
    float acc = 0.f;
    const float* Bp = &sBm[p][np * 32];
    const float* Cp = &sCm[p][np * 32];
#pragma unroll
    for (int j = 0; j < 32; ++j) {
      float stn = fmaf(dA, st[j], dtx * Bp[j]);
      st[j] = stn;
      acc = fmaf(stn, Cp[j], acc);
    }
    // reduce the 4 n-partitions (lanes 4d..4d+3, same wave32)
    acc += __shfl_xor(acc, 1, 32);
    acc += __shfl_xor(acc, 2, 32);
    if (np == 0)
      out[(rowbase + s) * INTER_ + h * 64 + d] = acc + Dh * xd;

    __syncthreads();
    p ^= 1;
  }
#undef SCAN_STAGE
}

// ---------------------------------------------------------------------------
// Gated RMSNorm: g = scan * silu(gate); out_bf16 = g * rsqrt(mean(g^2)+eps) * w
// One block (256 thr) per token row; 16 cols/thread
// ---------------------------------------------------------------------------
__global__ __launch_bounds__(256)
void gated_norm_kernel(const float* __restrict__ scan, const float* __restrict__ yall,
                       const float* __restrict__ nw, bf16_t* __restrict__ og) {
  const int r = blockIdx.x;
  const int t = threadIdx.x;
  const float* sr = scan + (size_t)r * INTER_;
  const float* gr = yall + (size_t)r * NTOT_;  // gate = cols [0,4096)
  float gv[16];
  float ss = 0.f;
#pragma unroll
  for (int i = 0; i < 16; ++i) {
    int c = t + i * 256;
    float z = gr[c];
    float v = sr[c] * (z / (1.f + __expf(-z)));
    gv[i] = v;
    ss = fmaf(v, v, ss);
  }
  ss += __shfl_xor(ss, 1, 32);
  ss += __shfl_xor(ss, 2, 32);
  ss += __shfl_xor(ss, 4, 32);
  ss += __shfl_xor(ss, 8, 32);
  ss += __shfl_xor(ss, 16, 32);
  __shared__ float red[8];
  if ((t & 31) == 0) red[t >> 5] = ss;
  __syncthreads();
  float tot = 0.f;
#pragma unroll
  for (int wv = 0; wv < 8; ++wv) tot += red[wv];
  float scale = rsqrtf(tot * (1.0f / INTER_) + 1e-5f);
#pragma unroll
  for (int i = 0; i < 16; ++i) {
    int c = t + i * 256;
    og[(size_t)r * INTER_ + c] = f2bf(gv[i] * scale * nw[c]);
  }
}

// ---------------------------------------------------------------------------
// Host launcher
// ---------------------------------------------------------------------------
extern "C" void kernel_launch(void* const* d_in, const int* in_sizes, int n_in,
                              void* d_out, int out_size, void* d_ws, size_t ws_size,
                              hipStream_t stream) {
  (void)in_sizes; (void)n_in; (void)out_size; (void)ws_size;
  const float* hs    = (const float*)d_in[0];
  const float* Wz    = (const float*)d_in[1];
  const float* Wx    = (const float*)d_in[2];
  const float* WBp   = (const float*)d_in[3];
  const float* WCp   = (const float*)d_in[4];
  const float* Wdt   = (const float*)d_in[5];
  const float* convw = (const float*)d_in[6];
  const float* convb = (const float*)d_in[7];
  const float* dtb   = (const float*)d_in[8];
  const float* Alog  = (const float*)d_in[9];
  const float* Dv    = (const float*)d_in[10];
  const float* normw = (const float*)d_in[11];
  const float* Wout  = (const float*)d_in[12];
  float* out = (float*)d_out;

  char* w = (char*)d_ws;
  size_t off = 0;
  auto take = [&](size_t bytes) -> char* {
    char* p = w + off;
    off = (off + bytes + 255) & ~(size_t)255;
    return p;
  };
  bf16_t* hsb   = (bf16_t*)take((size_t)MROWS_ * HIDDEN_ * 2);
  bf16_t* WallT = (bf16_t*)take((size_t)NTOT_ * HIDDEN_ * 2);
  float*  Yall  = (float*) take((size_t)MROWS_ * NTOT_ * 4);
  float*  xbc   = (float*) take((size_t)MROWS_ * CONVD_ * 4);
  float*  yscan = (float*) take((size_t)MROWS_ * INTER_ * 4);
  bf16_t* gbf   = (bf16_t*)take((size_t)MROWS_ * INTER_ * 2);
  bf16_t* WoutT = (bf16_t*)take((size_t)HIDDEN_ * INTER_ * 2);

  // 1) activations -> bf16
  cast_bf16_kernel<<<(MROWS_ * HIDDEN_) / 256, 256, 0, stream>>>(
      hs, hsb, MROWS_ * HIDDEN_);

  // 2) pack transposed bf16 weights: WallT = [gate|x|B|C|dt]^T, WoutT = Wout^T
  dim3 tb(32, 8);
  transpose_pack_kernel<<<dim3(INTER_ / 32, HIDDEN_ / 32), tb, 0, stream>>>(
      Wz, WallT, HIDDEN_, INTER_, 0, HIDDEN_);
  transpose_pack_kernel<<<dim3(INTER_ / 32, HIDDEN_ / 32), tb, 0, stream>>>(
      Wx, WallT, HIDDEN_, INTER_, INTER_, HIDDEN_);
  transpose_pack_kernel<<<dim3(GN_ / 32, HIDDEN_ / 32), tb, 0, stream>>>(
      WBp, WallT, HIDDEN_, GN_, 2 * INTER_, HIDDEN_);
  transpose_pack_kernel<<<dim3(GN_ / 32, HIDDEN_ / 32), tb, 0, stream>>>(
      WCp, WallT, HIDDEN_, GN_, 2 * INTER_ + GN_, HIDDEN_);
  transpose_pack_kernel<<<dim3(NHEADS_ / 32, HIDDEN_ / 32), tb, 0, stream>>>(
      Wdt, WallT, HIDDEN_, NHEADS_, DT_OFF_, HIDDEN_);
  transpose_pack_kernel<<<dim3(HIDDEN_ / 32, INTER_ / 32), tb, 0, stream>>>(
      Wout, WoutT, INTER_, HIDDEN_, 0, INTER_);

  // 3) fused input-projection GEMM -> Y_all [4096, 10304]
  wmma_gemm_bf16_kernel<<<dim3((NTOT_ + BN_ - 1) / BN_, MROWS_ / BM_), 256, 0, stream>>>(
      hsb, WallT, Yall, MROWS_, NTOT_, HIDDEN_);

  // 4) causal conv + silu on xBC slice
  conv_silu_kernel<<<(MROWS_ * CONVD_) / 256, 256, 0, stream>>>(
      Yall, convw, convb, xbc);

  // 5) selective scan
  ssm_scan_kernel<<<BATCH_ * NHEADS_, 256, 0, stream>>>(
      xbc, Yall, dtb, Alog, Dv, yscan);

  // 6) gated RMSNorm -> bf16
  gated_norm_kernel<<<MROWS_, 256, 0, stream>>>(yscan, Yall, normw, gbf);

  // 7) output projection GEMM -> d_out [4096, 2048] f32
  wmma_gemm_bf16_kernel<<<dim3(HIDDEN_ / BN_, MROWS_ / BM_), 256, 0, stream>>>(
      gbf, WoutT, out, MROWS_, HIDDEN_, INTER_);
}
